// EGNA_18880676233902
// MI455X (gfx1250) — matrix operand order repaired and checked by reading, not compiled, not run on hardware
//
#include <hip/hip_runtime.h>
#include <hip/hip_bf16.h>

// ---------------------------------------------------------------------------
// Types for CDNA5 WMMA
// ---------------------------------------------------------------------------
typedef __attribute__((ext_vector_type(16))) __bf16 bf16x16;
typedef __attribute__((ext_vector_type(8)))  __bf16 bf16x8;
typedef __attribute__((ext_vector_type(8)))  float  f32x8;

__device__ __forceinline__ unsigned short f2bf(float f) {
    union { float f; unsigned int u; } v; v.f = f;
    unsigned int u = v.u;
    unsigned int r = (u + 0x7FFFu + ((u >> 16) & 1u)) >> 16; // RNE
    return (unsigned short)r;
}

// ---------------------------------------------------------------------------
// Tiled GEMM: C[M,N] (+)= op(A)[M,K] @ B[K,N]
//   A: f32 or bf16 (ushort) storage, B/C: f32, bf16 WMMA math, f32 accumulate
// Block tile 128x128, K-step 32, LDS double-buffered + register-staged
// software pipeline: global loads for tile t+1 are in flight while tile t's
// WMMAs execute; the loadcnt wait lands after the math.
// 512 threads = 16 waves (4x4); each wave owns a 32x32 C tile (4 WMMA/step).
// ---------------------------------------------------------------------------
#define BM 128
#define BN 128
#define BK 32
#define LDA_S 40   // LDS row stride (bf16 units): 80B rows -> 16B/8B alignment
#define LDB_S 40

struct StageRegs {
    float4 b0, b1;   // B tile halves (always f32)
    float4 a0, a1;   // A tile (f32 source)
    uint4  au;       // A tile (bf16 source, 8 elements)
};

template<typename TA, bool TRANSA>
__device__ __forceinline__
void gload_fast(StageRegs& r,
                const TA* __restrict__ A, int lda,
                const float* __restrict__ B, int ldb,
                int m0, int n0, int k0, int tid)
{
    constexpr bool ABF = (sizeof(TA) == 2);
    const int bk0 = tid >> 5,          bn0 = (tid & 31) * 4;
    const int bk1 = (tid + 512) >> 5,  bn1 = ((tid + 512) & 31) * 4;
    r.b0 = *(const float4*)&B[(size_t)(k0 + bk0) * ldb + (n0 + bn0)];
    r.b1 = *(const float4*)&B[(size_t)(k0 + bk1) * ldb + (n0 + bn1)];
    if constexpr (ABF) {
        if constexpr (TRANSA) {
            const int k = tid >> 4, m8 = (tid & 15) * 8;
            r.au = *(const uint4*)&A[(size_t)(k0 + k) * lda + (m0 + m8)];
        } else {
            const int rr = tid >> 2, c8 = (tid & 3) * 8;
            r.au = *(const uint4*)&A[(size_t)(m0 + rr) * lda + (k0 + c8)];
        }
    } else {
        if constexpr (TRANSA) {
            const int k0i = tid >> 5,         m40 = (tid & 31) * 4;
            const int k1i = (tid + 512) >> 5, m41 = ((tid + 512) & 31) * 4;
            r.a0 = *(const float4*)&A[(size_t)(k0 + k0i) * lda + (m0 + m40)];
            r.a1 = *(const float4*)&A[(size_t)(k0 + k1i) * lda + (m0 + m41)];
        } else {
            const int r0 = tid >> 3,         c40 = (tid & 7) * 4;
            const int r1 = (tid + 512) >> 3, c41 = ((tid + 512) & 7) * 4;
            r.a0 = *(const float4*)&A[(size_t)(m0 + r0) * lda + (k0 + c40)];
            r.a1 = *(const float4*)&A[(size_t)(m0 + r1) * lda + (k0 + c41)];
        }
    }
}

template<typename TA, bool TRANSA>
__device__ __forceinline__
void store_fast(unsigned short* lA, unsigned short* lB,
                const StageRegs& r, int tid)
{
    constexpr bool ABF = (sizeof(TA) == 2);
    if constexpr (ABF) {
        if constexpr (TRANSA) {
            const int k = tid >> 4, m8 = (tid & 15) * 8;
            union { uint4 u; unsigned short s[8]; } va; va.u = r.au;
#pragma unroll
            for (int j = 0; j < 8; ++j) lA[(m8 + j) * LDA_S + k] = va.s[j];
        } else {
            const int rr = tid >> 2, c8 = (tid & 3) * 8;
            *(uint4*)&lA[rr * LDA_S + c8] = r.au;
        }
    } else {
        if constexpr (TRANSA) {
            const int k0i = tid >> 5,         m40 = (tid & 31) * 4;
            const int k1i = (tid + 512) >> 5, m41 = ((tid + 512) & 31) * 4;
            lA[(m40 + 0) * LDA_S + k0i] = f2bf(r.a0.x);
            lA[(m40 + 1) * LDA_S + k0i] = f2bf(r.a0.y);
            lA[(m40 + 2) * LDA_S + k0i] = f2bf(r.a0.z);
            lA[(m40 + 3) * LDA_S + k0i] = f2bf(r.a0.w);
            lA[(m41 + 0) * LDA_S + k1i] = f2bf(r.a1.x);
            lA[(m41 + 1) * LDA_S + k1i] = f2bf(r.a1.y);
            lA[(m41 + 2) * LDA_S + k1i] = f2bf(r.a1.z);
            lA[(m41 + 3) * LDA_S + k1i] = f2bf(r.a1.w);
        } else {
            const int r0 = tid >> 3,         c40 = (tid & 7) * 4;
            const int r1 = (tid + 512) >> 3, c41 = ((tid + 512) & 7) * 4;
            ushort4 p0, p1;
            p0.x = f2bf(r.a0.x); p0.y = f2bf(r.a0.y); p0.z = f2bf(r.a0.z); p0.w = f2bf(r.a0.w);
            p1.x = f2bf(r.a1.x); p1.y = f2bf(r.a1.y); p1.z = f2bf(r.a1.z); p1.w = f2bf(r.a1.w);
            *(ushort4*)&lA[r0 * LDA_S + c40] = p0;
            *(ushort4*)&lA[r1 * LDA_S + c41] = p1;
        }
    }
    const int bk0 = tid >> 5,          bn0 = (tid & 31) * 4;
    const int bk1 = (tid + 512) >> 5,  bn1 = ((tid + 512) & 31) * 4;
    lB[(bn0 + 0) * LDB_S + bk0] = f2bf(r.b0.x);
    lB[(bn0 + 1) * LDB_S + bk0] = f2bf(r.b0.y);
    lB[(bn0 + 2) * LDB_S + bk0] = f2bf(r.b0.z);
    lB[(bn0 + 3) * LDB_S + bk0] = f2bf(r.b0.w);
    lB[(bn1 + 0) * LDB_S + bk1] = f2bf(r.b1.x);
    lB[(bn1 + 1) * LDB_S + bk1] = f2bf(r.b1.y);
    lB[(bn1 + 2) * LDB_S + bk1] = f2bf(r.b1.z);
    lB[(bn1 + 3) * LDB_S + bk1] = f2bf(r.b1.w);
}

template<typename TA, bool TRANSA>
__device__ __forceinline__
void load_tiles_guard(unsigned short* lA, unsigned short* lB,
                      const TA* __restrict__ A, int lda,
                      const float* __restrict__ B, int ldb,
                      int m0, int n0, int k0, int M, int N, int K, int tid)
{
    constexpr bool ABF = (sizeof(TA) == 2);
    if constexpr (ABF) {
        if constexpr (TRANSA) {
            const int k = tid >> 4, m8 = (tid & 15) * 8;
            const bool kok = (k0 + k) < K;
#pragma unroll
            for (int j = 0; j < 8; ++j) {
                unsigned short s = 0;
                if (kok && (m0 + m8 + j) < M) s = A[(size_t)(k0 + k) * lda + (m0 + m8 + j)];
                lA[(m8 + j) * LDA_S + k] = s;
            }
        } else {
            const int r = tid >> 2, c8 = (tid & 3) * 8;
            const bool mok = (m0 + r) < M;
#pragma unroll
            for (int j = 0; j < 8; ++j) {
                unsigned short s = 0;
                if (mok && (k0 + c8 + j) < K) s = A[(size_t)(m0 + r) * lda + (k0 + c8 + j)];
                lA[r * LDA_S + c8 + j] = s;
            }
        }
    } else {
#pragma unroll
        for (int i = 0; i < 2; ++i) {
            const int idx = tid + i * 512;
            if constexpr (TRANSA) {
                const int k = idx >> 5, m4 = (idx & 31) * 4;
                const bool kok = (k0 + k) < K;
#pragma unroll
                for (int j = 0; j < 4; ++j) {
                    float f = 0.f;
                    if (kok && (m0 + m4 + j) < M) f = A[(size_t)(k0 + k) * lda + (m0 + m4 + j)];
                    lA[(m4 + j) * LDA_S + k] = f2bf(f);
                }
            } else {
                const int r = idx >> 3, c4 = (idx & 7) * 4;
                const bool mok = (m0 + r) < M;
#pragma unroll
                for (int j = 0; j < 4; ++j) {
                    float f = 0.f;
                    if (mok && (k0 + c4 + j) < K) f = A[(size_t)(m0 + r) * lda + (k0 + c4 + j)];
                    lA[r * LDA_S + c4 + j] = f2bf(f);
                }
            }
        }
    }
#pragma unroll
    for (int i = 0; i < 2; ++i) {
        const int idx = tid + i * 512;
        const int k = idx >> 5, n4 = (idx & 31) * 4;
        const bool kok = (k0 + k) < K;
#pragma unroll
        for (int j = 0; j < 4; ++j) {
            float f = 0.f;
            if (kok && (n0 + n4 + j) < N) f = B[(size_t)(k0 + k) * ldb + (n0 + n4 + j)];
            lB[(n4 + j) * LDB_S + k] = f2bf(f);
        }
    }
}

template<typename TA, bool TRANSA, bool ACC>
__global__ __launch_bounds__(512)
void gemm_wmma_kernel(const TA* __restrict__ A, int lda,
                      const float* __restrict__ B, int ldb,
                      float* __restrict__ C, int ldc,
                      int M, int N, int K)
{
    __shared__ __align__(16) unsigned short lA[2][BM * LDA_S];
    __shared__ __align__(16) unsigned short lB[2][BN * LDB_S];

    const int tid  = threadIdx.x;
    const int lane = tid & 31;
    const int wave = tid >> 5;    // 0..15
    const int wm   = wave >> 2;   // 0..3
    const int wn   = wave & 3;    // 0..3
    const int m0   = blockIdx.y * BM;
    const int n0   = blockIdx.x * BN;

    const int half = lane >> 4;   // 0/1
    const int l16  = lane & 15;

    f32x8 acc[2][2] = {};
    if (ACC) {
#pragma unroll
        for (int i = 0; i < 2; ++i)
#pragma unroll
            for (int j = 0; j < 2; ++j) {
                const int nn = n0 + wn * 32 + j * 16 + l16;
#pragma unroll
                for (int v = 0; v < 8; ++v) {
                    const int mm = m0 + wm * 32 + i * 16 + v + 8 * half;
                    float c = 0.f;
                    if (mm < M && nn < N) c = C[(size_t)mm * ldc + nn];
                    acc[i][j][v] = c;
                }
            }
    }

    const bool edge = (m0 + BM > M) || (n0 + BN > N);
    const int  KT   = (K + BK - 1) / BK;

    StageRegs rg;
    auto stage_load = [&](int t) -> bool {
        const int kb = t * BK;
        const bool fast = !edge && kb + BK <= K;
        if (fast) gload_fast<TA, TRANSA>(rg, A, lda, B, ldb, m0, n0, kb, tid);
        return fast;
    };
    auto stage_store = [&](int t, bool fast, int buf) {
        if (fast) store_fast<TA, TRANSA>(lA[buf], lB[buf], rg, tid);
        else      load_tiles_guard<TA, TRANSA>(lA[buf], lB[buf], A, lda, B, ldb,
                                               m0, n0, t * BK, M, N, K, tid);
    };
    auto mma_from = [&](int buf) {
        const unsigned short* pA = lA[buf];
        const unsigned short* pB = lB[buf];
        union FR { bf16x16 v; bf16x8 h[2]; };
        FR af[2];
#pragma unroll
        for (int i = 0; i < 2; ++i) {
            const int ml = wm * 32 + i * 16 + l16;
            af[i].h[0] = *(const bf16x8*)&pA[ml * LDA_S + half * 8];
            af[i].h[1] = *(const bf16x8*)&pA[ml * LDA_S + 16 + half * 8];
        }
        FR bfr[2];
#pragma unroll
        for (int j = 0; j < 2; ++j) {
            const int nl = wn * 32 + j * 16 + l16;
            bfr[j].h[0] = *(const bf16x8*)&pB[nl * LDB_S + half * 8];
            bfr[j].h[1] = *(const bf16x8*)&pB[nl * LDB_S + 16 + half * 8];
        }
#pragma unroll
        for (int i = 0; i < 2; ++i)
#pragma unroll
            for (int j = 0; j < 2; ++j)
                acc[i][j] = __builtin_amdgcn_wmma_f32_16x16x32_bf16(
                    false, af[i].v, false, bfr[j].v, (short)0, acc[i][j], false, false);
    };

    // ---- software pipeline: regs hold tile t while LDS holds tile t-1 ----
    bool fcur = stage_load(0);
    stage_store(0, fcur, 0);
    bool fnext = (KT > 1) ? stage_load(1) : false;
    __syncthreads();

    int cur = 0;
    for (int t = 1; t < KT; ++t) {
        mma_from(cur);                    // compute tile t-1 (loads for t in flight)
        stage_store(t, fnext, cur ^ 1);   // loadcnt wait lands here, after the math
        if (t + 1 < KT) {
            fnext = stage_load(t + 1);    // issue next loads before the barrier
            if (!edge && (t + 2) * BK + BK <= K) {   // stream tile t+2 toward L2
                if (TRANSA)
                    __builtin_prefetch(&A[(size_t)((t + 2) * BK + (tid & 31)) * lda + m0], 0, 1);
                else
                    __builtin_prefetch(&A[(size_t)(m0 + (tid & 127)) * lda + (t + 2) * BK], 0, 1);
                __builtin_prefetch(&B[(size_t)((t + 2) * BK + (tid & 31)) * ldb + n0], 0, 1);
            }
        }
        __syncthreads();
        cur ^= 1;
    }
    mma_from(cur);                        // last tile

#pragma unroll
    for (int i = 0; i < 2; ++i)
#pragma unroll
        for (int j = 0; j < 2; ++j) {
            const int nn = n0 + wn * 32 + j * 16 + l16;
#pragma unroll
            for (int v = 0; v < 8; ++v) {
                const int mm = m0 + wm * 32 + i * 16 + v + 8 * half;
                if (mm < M && nn < N) C[(size_t)mm * ldc + nn] = acc[i][j][v];
            }
        }
}

// ---------------------------------------------------------------------------
// Elementwise / reduction helper kernels
// ---------------------------------------------------------------------------
__global__ void zero_kernel(float* p, int n) {
    int i = blockIdx.x * 256 + threadIdx.x;
    if (i < n) p[i] = 0.f;
}

__global__ void colstat_kernel(const float* __restrict__ x, int M, int N,
                               float* __restrict__ st) {
    int c = blockIdx.x * blockDim.x + threadIdx.x;
    if (c >= N) return;
    int r0 = blockIdx.y * 512;
    int r1 = r0 + 512; if (r1 > M) r1 = M;
    float s = 0.f, q = 0.f;
    for (int r = r0; r < r1; ++r) {
        float v = x[(size_t)r * N + c];
        s += v; q += v * v;
    }
    atomicAdd(&st[c], s);
    atomicAdd(&st[N + c], q);
}

// mode: 0 = relu, 1 = leaky(0.01), 2 = none
__global__ void bn_apply_kernel(const float* __restrict__ x, float* __restrict__ y,
                                const float* __restrict__ st,
                                const float* __restrict__ g,
                                const float* __restrict__ be,
                                int M, int N, int mode) {
    int i = blockIdx.x * 256 + threadIdx.x;
    if (i >= M * N) return;
    int c = i % N;
    float mean = st[c] / (float)M;
    float var  = st[N + c] / (float)M - mean * mean;
    float v = (x[i] - mean) * rsqrtf(var + 1e-5f) * g[c] + be[c];
    if (mode == 0)      v = v > 0.f ? v : 0.f;
    else if (mode == 1) v = v > 0.f ? v : 0.01f * v;
    y[i] = v;
}

// powers written directly as bf16 (consumed only as WMMA A-operands)
__global__ void pl_pow_kernel(const float* __restrict__ pl,
                              unsigned short* __restrict__ out, long n) {
    long i = (long)blockIdx.x * 256 + threadIdx.x;
    if (i >= n) return;
    float x  = pl[i];
    float x3 = x * x * x;
    float x6 = x3 * x3;
    float x9 = x6 * x3;
    float x12 = x6 * x6;
    out[i]         = f2bf(x);
    out[n + i]     = f2bf(x3);
    out[2 * n + i] = f2bf(x6);
    out[3 * n + i] = f2bf(x9);
    out[4 * n + i] = f2bf(x12);
}

__global__ void gather_rows_kernel(const float* __restrict__ src,
                                   const int* __restrict__ idx,
                                   float* __restrict__ dst, int R, int N) {
    int i = blockIdx.x * 256 + threadIdx.x;
    if (i >= R * N) return;
    int r = i / N, c = i % N;
    dst[i] = src[(size_t)idx[r] * N + c];
}

// out[r, coff + d] = sum_e w[e] * buf[e][r][d] + wb
__global__ void wsum_kernel(const float* __restrict__ buf,
                            const float* __restrict__ w,
                            const float* __restrict__ wb,
                            float* __restrict__ dst,
                            int NR, int D, int ldd, int coff) {
    int i = blockIdx.x * 256 + threadIdx.x;
    if (i >= NR * D) return;
    int r = i / D, d = i % D;
    float s = wb[0];
    size_t plane = (size_t)NR * D;
#pragma unroll
    for (int e = 0; e < 5; ++e) s += w[e] * buf[plane * e + i];
    dst[(size_t)r * ldd + coff + d] = s;
}

// segment max over sorted graph indices; one thread per column
__global__ void segmax_kernel(const float* __restrict__ x,
                              const int* __restrict__ gidx,
                              float* __restrict__ dst,
                              int R, int N, int ldd, int coff) {
    int c = blockIdx.x * blockDim.x + threadIdx.x;
    if (c >= N) return;
    for (int g = 0; g < 32; ++g) dst[(size_t)g * ldd + coff + c] = -3.4e38f;
    float cur = -3.4e38f;
    int cg = gidx[0];
    for (int r = 0; r < R; ++r) {
        int g = gidx[r];
        if (g != cg) {
            dst[(size_t)cg * ldd + coff + c] = cur;
            cg = g; cur = -3.4e38f;
        }
        float v = x[(size_t)r * N + c];
        cur = v > cur ? v : cur;
    }
    dst[(size_t)cg * ldd + coff + c] = cur;
}

__global__ void outdot_kernel(const float* __restrict__ q,
                              const float* __restrict__ w,
                              const float* __restrict__ b,
                              float* __restrict__ out) {
    __shared__ float red[256];
    int g = blockIdx.x, t = threadIdx.x;
    float s = 0.f;
    for (int d = t; d < 512; d += 256) s += q[(size_t)g * 512 + d] * w[d];
    red[t] = s; __syncthreads();
    for (int o = 128; o > 0; o >>= 1) {
        if (t < o) red[t] += red[t + o];
        __syncthreads();
    }
    if (t == 0) out[g] = red[0] + b[0];
}

// ---------------------------------------------------------------------------
// Host side
// ---------------------------------------------------------------------------
static inline void gemm_f32(hipStream_t s, const float* A, int lda,
                            const float* B, int ldb, float* C, int ldc,
                            int M, int N, int K, bool acc) {
    dim3 g((N + BN - 1) / BN, (M + BM - 1) / BM), b(512);
    if (acc) gemm_wmma_kernel<float, false, true ><<<g, b, 0, s>>>(A, lda, B, ldb, C, ldc, M, N, K);
    else     gemm_wmma_kernel<float, false, false><<<g, b, 0, s>>>(A, lda, B, ldb, C, ldc, M, N, K);
}

static inline void gemm_bfA(hipStream_t s, const unsigned short* A, int lda, bool ta,
                            const float* B, int ldb, float* C, int ldc,
                            int M, int N, int K) {
    dim3 g((N + BN - 1) / BN, (M + BM - 1) / BM), b(512);
    if (ta) gemm_wmma_kernel<unsigned short, true,  false><<<g, b, 0, s>>>(A, lda, B, ldb, C, ldc, M, N, K);
    else    gemm_wmma_kernel<unsigned short, false, false><<<g, b, 0, s>>>(A, lda, B, ldb, C, ldc, M, N, K);
}

static inline void run_bn(hipStream_t s, const float* x, float* y, int M, int N,
                          const float* g, const float* be, int mode, float* st) {
    int n2 = 2 * N;
    zero_kernel<<<(n2 + 255) / 256, 256, 0, s>>>(st, n2);
    dim3 gs((N + 255) / 256, (M + 511) / 512);
    colstat_kernel<<<gs, 256, 0, s>>>(x, M, N, st);
    int tot = M * N;
    bn_apply_kernel<<<(tot + 255) / 256, 256, 0, s>>>(x, y, st, g, be, M, N, mode);
}

extern "C" void kernel_launch(void* const* d_in, const int* in_sizes, int n_in,
                              void* d_out, int out_size, void* d_ws, size_t ws_size,
                              hipStream_t stream) {
    (void)in_sizes; (void)n_in; (void)out_size; (void)ws_size;
    const int Np = 8000, Nr = 3000, Nl = 3000;

    const float* px  = (const float*)d_in[0];
    const float* lx  = (const float*)d_in[1];
    const float* pp  = (const float*)d_in[2];
    const float* pl  = (const float*)d_in[3];
    const float* ll  = (const float*)d_in[4];
    const float* ipW = (const float*)d_in[5];
    const float* ipG = (const float*)d_in[7];
    const float* ipBe= (const float*)d_in[8];
    const float* ilW = (const float*)d_in[9];
    const float* ilG = (const float*)d_in[11];
    const float* ilBe= (const float*)d_in[12];
    const float* pgW = (const float*)d_in[13];
    const float* pgG = (const float*)d_in[15];
    const float* pgBe= (const float*)d_in[16];
    const float* lgW = (const float*)d_in[17];
    const float* lgG = (const float*)d_in[19];
    const float* lgBe= (const float*)d_in[20];
    const float* e1_lpW = (const float*)d_in[21];
    const float* e1_lpG = (const float*)d_in[23];
    const float* e1_lpBe= (const float*)d_in[24];
    const float* e1_plW = (const float*)d_in[25];
    const float* e1_plG = (const float*)d_in[27];
    const float* e1_plBe= (const float*)d_in[28];
    const float* e1_wp  = (const float*)d_in[29];
    const float* e1_wpb = (const float*)d_in[30];
    const float* e1_wl  = (const float*)d_in[31];
    const float* e1_wlb = (const float*)d_in[32];
    const float* e2_lpW = (const float*)d_in[33];
    const float* e2_lpG = (const float*)d_in[35];
    const float* e2_lpBe= (const float*)d_in[36];
    const float* e2_plW = (const float*)d_in[37];
    const float* e2_plG = (const float*)d_in[39];
    const float* e2_plBe= (const float*)d_in[40];
    const float* e2_wp  = (const float*)d_in[41];
    const float* e2_wpb = (const float*)d_in[42];
    const float* e2_wl  = (const float*)d_in[43];
    const float* e2_wlb = (const float*)d_in[44];
    const float* emW  = (const float*)d_in[45];
    const float* emG  = (const float*)d_in[47];
    const float* emBe = (const float*)d_in[48];
    const float* outW = (const float*)d_in[49];
    const float* outB = (const float*)d_in[50];
    const int* res_idx = (const int*)d_in[51];
    const int* prot_g  = (const int*)d_in[52];
    const int* lig_g   = (const int*)d_in[53];

    // --- workspace bump allocator (float units) ---
    float* w = (float*)d_ws;
    size_t off = 0;
    auto alloc = [&](size_t n) { float* p = w + off; off += n; return p; };

    const long PLN = (long)Nr * Nl;                       // 9,000,000
    unsigned short* pl_pows = (unsigned short*)alloc((5 * (size_t)PLN + 1) / 2); // bf16
    float* tp   = alloc((size_t)Np * 128);
    float* tl   = alloc((size_t)Nl * 128);
    float* px1  = alloc((size_t)Np * 128);
    float* lx1  = alloc((size_t)Nl * 128);
    float* px2  = alloc((size_t)Np * 256);
    float* lx2  = alloc((size_t)Nl * 256);
    float* zbuf = alloc((size_t)Np * 256);          // >= 3000*512 too
    float* ix0  = alloc((size_t)Nr * 256);
    float* hbuf = alloc((size_t)Nr * 512);
    float* bufP = alloc(5 * (size_t)Nr * 512);      // px_p[e]
    float* bufL = alloc(5 * (size_t)Nr * 512);      // lx_p[e]
    float* ix_cat = alloc((size_t)Nr * 1024);
    float* lx_cat = alloc((size_t)Nr * 1024);
    float* ilx  = alloc((size_t)32 * 2048);
    float* qpre = alloc((size_t)32 * 512);
    float* qbuf = alloc((size_t)32 * 512);
    float* st   = alloc(4096);

    // 1) adjacency powers (bf16)
    pl_pow_kernel<<<(unsigned)((PLN + 255) / 256), 256, 0, stream>>>(pl, pl_pows, PLN);

    // 2) px1 = relu(BN(px @ ipW))     (linear biases cancel exactly under BN)
    gemm_f32(stream, px, 512, ipW, 128, tp, 128, Np, 128, 512, false);
    run_bn(stream, tp, px1, Np, 128, ipG, ipBe, 0, st);

    // 3) lx1 = relu(BN(lx @ ilW))
    gemm_f32(stream, lx, 128, ilW, 128, tl, 128, Nl, 128, 128, false);
    run_bn(stream, tl, lx1, Nl, 128, ilG, ilBe, 0, st);

    // 4) px2 = relu(BN((pp @ px1) @ pgW))
    gemm_f32(stream, pp, Np, px1, 128, tp, 128, Np, 128, Np, false);
    gemm_f32(stream, tp, 128, pgW, 256, zbuf, 256, Np, 256, 128, false);
    run_bn(stream, zbuf, px2, Np, 256, pgG, pgBe, 0, st);

    // 5) lx2 = relu(BN((ll @ lx1) @ lgW))
    gemm_f32(stream, ll, Nl, lx1, 128, tl, 128, Nl, 128, Nl, false);
    gemm_f32(stream, tl, 128, lgW, 256, zbuf, 256, Nl, 256, 128, false);
    run_bn(stream, zbuf, lx2, Nl, 256, lgG, lgBe, 0, st);

    // 6) ix0 = px2[res_idx]
    gather_rows_kernel<<<(Nr * 256 + 255) / 256, 256, 0, stream>>>(px2, res_idx, ix0, Nr, 256);

    // exchange helper: out = relu(BN(concat(op(am)@x1, x2) @ W))
    auto exchange = [&](const unsigned short* am, bool ta,
                        const float* x1, int ld1, int d1,
                        const float* x2, int ld2, int d2,
                        const float* W, const float* G, const float* Be,
                        float* outbuf) {
        gemm_bfA(stream, am, Nl, ta, x1, ld1, hbuf, d1, Nr, d1, Nl);
        gemm_f32(stream, hbuf, d1, W, 512, zbuf, 512, Nr, 512, d1, false);
        gemm_f32(stream, x2, ld2, W + (size_t)d1 * 512, 512, zbuf, 512, Nr, 512, d2, true);
        run_bn(stream, zbuf, outbuf, Nr, 512, G, Be, 0, st);
    };

    const size_t PL = (size_t)Nr * 512;

    // 7) EIRL layer 1 (din = 512)
    for (int e = 0; e < 5; ++e) {
        const unsigned short* ple = pl_pows + (size_t)e * PLN;
        exchange(ple, true,  ix0, 256, 256, lx2, 256, 256,
                 e1_lpW + (size_t)e * 512 * 512, e1_lpG + e * 512, e1_lpBe + e * 512,
                 bufP + e * PL);
        exchange(ple, false, lx2, 256, 256, ix0, 256, 256,
                 e1_plW + (size_t)e * 512 * 512, e1_plG + e * 512, e1_plBe + e * 512,
                 bufL + e * PL);
    }
    // ix_g = sum_e wp[e]*lx_p[e] + wpb (note the swap in the reference)
    wsum_kernel<<<(Nr * 512 + 255) / 256, 256, 0, stream>>>(bufL, e1_wp, e1_wpb, ix_cat, Nr, 512, 1024, 0);
    wsum_kernel<<<(Nr * 512 + 255) / 256, 256, 0, stream>>>(bufP, e1_wl, e1_wlb, lx_cat, Nr, 512, 1024, 0);

    // 8) EIRL layer 2 (din = 1024) — x1/x2 read from the cat buffers (cols 0..511)
    for (int e = 0; e < 5; ++e) {
        const unsigned short* ple = pl_pows + (size_t)e * PLN;
        exchange(ple, true,  ix_cat, 1024, 512, lx_cat, 1024, 512,
                 e2_lpW + (size_t)e * 1024 * 512, e2_lpG + e * 512, e2_lpBe + e * 512,
                 bufP + e * PL);
        exchange(ple, false, lx_cat, 1024, 512, ix_cat, 1024, 512,
                 e2_plW + (size_t)e * 1024 * 512, e2_plG + e * 512, e2_plBe + e * 512,
                 bufL + e * PL);
    }
    wsum_kernel<<<(Nr * 512 + 255) / 256, 256, 0, stream>>>(bufL, e2_wp, e2_wpb, ix_cat, Nr, 512, 1024, 512);
    wsum_kernel<<<(Nr * 512 + 255) / 256, 256, 0, stream>>>(bufP, e2_wl, e2_wlb, lx_cat, Nr, 512, 1024, 512);

    // 9) segment max -> ilx (32 x 2048)
    segmax_kernel<<<(1024 + 255) / 256, 256, 0, stream>>>(ix_cat, prot_g, ilx, Nr, 1024, 2048, 0);
    segmax_kernel<<<(1024 + 255) / 256, 256, 0, stream>>>(lx_cat, lig_g,  ilx, Nl, 1024, 2048, 1024);

    // 10) q = leaky(BN(ilx @ emW)); out = q @ outW + outB
    gemm_f32(stream, ilx, 2048, emW, 512, qpre, 512, 32, 512, 2048, false);
    run_bn(stream, qpre, qbuf, 32, 512, emG, emBe, 1, st);
    outdot_kernel<<<32, 256, 0, stream>>>(qbuf, outW, outB, (float*)d_out);
}
